// MultiHeadAttention4AKT_21320217657941
// MI455X (gfx1250) — matrix operand order repaired and checked by reading, not compiled, not run on hardware
//
#include <hip/hip_runtime.h>
#include <hip/hip_bf16.h>
#include <math.h>

// Problem constants (match reference)
#define B_  32
#define S_  512
#define D_  512
#define H_  8
#define DF_ 64

typedef __bf16 bf16;
typedef __attribute__((ext_vector_type(16))) __bf16 v16bf;
typedef __attribute__((ext_vector_type(8)))  float  v8f;

// ---------------------------------------------------------------------------
// WMMA fragment helpers (wave32, v_wmma_f32_16x16x32_bf16)
//
// A (16x32 bf16), k order chosen by this code: lanes 0-15 carry M=0..15 with K=0..15 in elems 0..15;
// lanes 16-31 carry K=16..31.
// => per lane: one contiguous 16-element chunk at K=(lane>>4)*16.
//
// B (32x16 bf16), same k order as A: lane (n=lane&15) holds column n; lanes 0-15 hold K=0..15,
// lanes 16-31 hold K=16..31 => one contiguous 16-element K-chunk per lane
// when the source is K-contiguous.
//
// C/D (16x16 f32): lane&15 = N, element r -> M = r + (lane>>4)*8.
// ---------------------------------------------------------------------------
template <typename T>
static __device__ inline v16bf load_frag_a(const T* __restrict__ A, int lda) {
  const int lane = threadIdx.x & 31;
  const int m    = lane & 15;
  const int kb   = (lane >> 4) * 16;
  const T* p = A + (size_t)m * lda;
  v16bf a;
#pragma unroll
  for (int i = 0; i < 16; ++i) a[i] = (bf16)p[kb + i];
  return a;
}

template <typename T>
static __device__ inline v16bf load_frag_b(const T* __restrict__ Bm, int ldb) {
  const int lane = threadIdx.x & 31;
  const int n    = lane & 15;
  const int kb   = (lane >> 4) * 16;
  const T* p = Bm + (size_t)n * ldb + kb;
  v16bf b;
#pragma unroll
  for (int i = 0; i < 16; ++i) b[i] = (bf16)p[i];
  return b;
}

static __device__ inline v8f wmma_bf16(v16bf a, v16bf b, v8f c) {
  return __builtin_amdgcn_wmma_f32_16x16x32_bf16(
      /*neg_a=*/false, a, /*neg_b=*/false, b,
      /*c_mod=*/(short)0, c, /*reuse_a=*/false, /*reuse_b=*/false);
}

// ---------------------------------------------------------------------------
// Generic GEMM: C[m,n] = sum_k X[m,k] * W[n,k] + bias[n]   (i.e. X @ W^T + b)
// One wave per 16x64 output stripe; A fragment reused across 4 B tiles.
// MODE 0: store bf16 as [B,H,S,DF]   (split-heads layout for qh / kh)
// MODE 1: store bf16 as [B,H,DF,S]   (transposed, for vh -> K-contiguous B)
// MODE 2: store f32 row-major [M,N]  (final output)
// ---------------------------------------------------------------------------
template <typename TX, int MODE>
__global__ __launch_bounds__(32) void gemm_xwT_wmma(
    const TX* __restrict__ X, const float* __restrict__ W,
    const float* __restrict__ bias, void* __restrict__ out,
    int M, int N, int K) {
  const int m0 = blockIdx.x * 16;
  const int n0 = blockIdx.y * 64;

  const v8f zero = {0.f, 0.f, 0.f, 0.f, 0.f, 0.f, 0.f, 0.f};
  v8f acc[4];
#pragma unroll
  for (int t = 0; t < 4; ++t) acc[t] = zero;

  for (int k0 = 0; k0 < K; k0 += 32) {
    v16bf a = load_frag_a(X + (size_t)m0 * K + k0, K);
#pragma unroll
    for (int t = 0; t < 4; ++t) {
      v16bf bfrag = load_frag_b(W + (size_t)(n0 + 16 * t) * K + k0, K);
      acc[t] = wmma_bf16(a, bfrag, acc[t]);
    }
  }

  const int lane = threadIdx.x & 31;
  const int n_l  = lane & 15;
  const int mh   = (lane >> 4) * 8;
#pragma unroll
  for (int t = 0; t < 4; ++t) {
#pragma unroll
    for (int r = 0; r < 8; ++r) {
      const int mg = m0 + mh + r;
      const int ng = n0 + 16 * t + n_l;
      const float val = acc[t][r] + bias[ng];
      if (MODE == 0) {
        const int bb = mg >> 9, s = mg & (S_ - 1);
        const int hh = ng >> 6, dd = ng & (DF_ - 1);
        ((bf16*)out)[(((size_t)bb * H_ + hh) * S_ + s) * DF_ + dd] = (bf16)val;
      } else if (MODE == 1) {
        const int bb = mg >> 9, s = mg & (S_ - 1);
        const int hh = ng >> 6, dd = ng & (DF_ - 1);
        ((bf16*)out)[(((size_t)bb * H_ + hh) * DF_ + dd) * S_ + s] = (bf16)val;
      } else {
        ((float*)out)[(size_t)mg * N + ng] = val;
      }
    }
  }
}

// ---------------------------------------------------------------------------
// Fused AKT attention: one wave per (b, h, 32-query-row stripe).
//  1) scores stripe [32 x 512] = (qh . kh^T)/8 via WMMA -> LDS (f32); the two
//     16-row tiles share each kh B-fragment (halves kh traffic).
//  2) per-row scan, ONE LANE PER ROW (all 32 lanes busy): softmax1 -> cumsum
//     -> distance decay -> rescale -> softmax2, native-exp transcendentals.
//  3) out stripe [32 x 64] = P . vh via WMMA, k-outer loop: each LDS prob
//     A-fragment pair is loaded once and used by all 4 DF tiles; each vhT
//     B-fragment is shared by both 16-row q-tiles (8 WMMAs per k-step).
// LDS: 32x513 f32 (conflict-free column walks) + 32x520 bf16 ~ 97 KB.
// ---------------------------------------------------------------------------
__global__ __launch_bounds__(32) void akt_attention_wmma(
    const bf16* __restrict__ qh, const bf16* __restrict__ kh,
    const bf16* __restrict__ vhT, const float* __restrict__ qde,
    const float* __restrict__ gammas, const int* __restrict__ zero_pad,
    bf16* __restrict__ attn) {
  __shared__ float sc[32][513];
  __shared__ bf16  pb[32][520];

  const int h    = blockIdx.y;
  const int b    = blockIdx.z;
  const int q0   = blockIdx.x * 32;
  const int lane = threadIdx.x;

  const size_t bh = (size_t)b * H_ + h;
  const bf16* qbase = qh + (bh * S_ + q0) * DF_;

  const int n_l = lane & 15;
  const int mh  = (lane >> 4) * 8;
  const v8f zero = {0.f, 0.f, 0.f, 0.f, 0.f, 0.f, 0.f, 0.f};
  const float scale = 0.125f;  // 1/sqrt(64)

  // ---- scores stripe: both q-tiles share the kh B-fragments ----
  {
    // A fragments for both 16-row query tiles (K = DF = 64 -> two 32 steps)
    const v16bf a00 = load_frag_a(qbase, DF_);
    const v16bf a01 = load_frag_a(qbase + 32, DF_);
    const v16bf a10 = load_frag_a(qbase + 16 * DF_, DF_);
    const v16bf a11 = load_frag_a(qbase + 16 * DF_ + 32, DF_);

    for (int n0 = 0; n0 < S_; n0 += 16) {
      const bf16* kbase = kh + (bh * S_ + n0) * DF_;
      v16bf b0 = load_frag_b(kbase, DF_);
      v16bf b1 = load_frag_b(kbase + 32, DF_);
      v8f c0 = zero, c1 = zero;
      c0 = wmma_bf16(a00, b0, c0);
      c0 = wmma_bf16(a01, b1, c0);
      c1 = wmma_bf16(a10, b0, c1);
      c1 = wmma_bf16(a11, b1, c1);
#pragma unroll
      for (int r = 0; r < 8; ++r) {
        sc[mh + r][n0 + n_l]      = c0[r] * scale;
        sc[16 + mh + r][n0 + n_l] = c1[r] * scale;
      }
    }
  }
  __syncthreads();

  // ---- per-row scan passes: one lane per query row, all 32 lanes ----
  {
    const int m  = lane;
    const int qg = q0 + m;
    const int L  = qg + 1;  // causal: keys 0..qg valid

    const float gm = gammas[h];
    const float sp = (gm > 20.f) ? gm : __logf(1.f + __expf(gm));
    const float g  = -sp;  // gamma = -softplus(gammas[h])
    const float* drow = qde + ((size_t)b * S_ + qg) * S_;

    // pass 1: online max/sum for softmax1
    float m1 = -3.0e38f, s1 = 0.f;
    for (int k2 = 0; k2 < L; ++k2) {
      const float x = sc[m][k2];
      if (x > m1) { s1 *= __expf(m1 - x); m1 = x; }
      s1 += __expf(x - m1);
    }
    const float inv_s1 = __builtin_amdgcn_rcpf(s1);

    // pass 2: cumsum -> distance decay -> rescaled scores + online max/sum
    float cum = 0.f, m2 = -3.0e38f, s2 = 0.f;
    for (int k2 = 0; k2 < L; ++k2) {
      const float x = sc[m][k2];
      const float p = __expf(x - m1) * inv_s1;
      cum += p;
      float rem = 1.f - cum;           // disttot - distcum (disttot == 1)
      if (rem < 0.f) rem = 0.f;
      const float pos  = (float)(qg - k2);
      const float dist = __builtin_amdgcn_sqrtf(rem * pos);
      const float sig  = __builtin_amdgcn_rcpf(1.f + __expf(-drow[k2]));
      const float diff = __expf(sig);
      float eff = __expf(dist * g * diff);
      eff = fminf(fmaxf(eff, 1e-5f), 1e5f);
      const float xn = x * eff;
      sc[m][k2] = xn;
      if (xn > m2) { s2 *= __expf(m2 - xn); m2 = xn; }
      s2 += __expf(xn - m2);
    }
    const float inv_s2 = __builtin_amdgcn_rcpf(s2);

    // pass 3: normalize, zero_pad row 0, zero masked tail, write bf16 probs
    const int  zp   = *zero_pad;
    const bool zrow = (zp != 0) && (qg == 0);
    for (int k2 = 0; k2 < L; ++k2) {
      const float p2 = __expf(sc[m][k2] - m2) * inv_s2;
      pb[m][k2] = zrow ? (bf16)0.f : (bf16)p2;
    }
    for (int k2 = L; k2 < S_; ++k2) pb[m][k2] = (bf16)0.f;
  }
  __syncthreads();

  // ---- out stripe = P @ V: k-outer, A-frags shared across 4 DF tiles,
  //      B-frags shared across both q-tiles (8 WMMAs per k-step) ----
  {
    v8f acc0[4], acc1[4];
#pragma unroll
    for (int t = 0; t < 4; ++t) { acc0[t] = zero; acc1[t] = zero; }

    for (int k0 = 0; k0 < S_; k0 += 32) {
      v16bf a0 = load_frag_a(&pb[0][k0], 520);
      v16bf a1 = load_frag_a(&pb[16][k0], 520);
#pragma unroll
      for (int t = 0; t < 4; ++t) {
        v16bf bv = load_frag_b(vhT + (bh * DF_ + 16 * t) * S_ + k0, S_);
        acc0[t] = wmma_bf16(a0, bv, acc0[t]);
        acc1[t] = wmma_bf16(a1, bv, acc1[t]);
      }
    }

#pragma unroll
    for (int t = 0; t < 4; ++t) {
#pragma unroll
      for (int r = 0; r < 8; ++r) {
        const size_t col = h * DF_ + 16 * t + n_l;
        attn[((size_t)b * S_ + q0 + mh + r) * D_ + col]      = (bf16)acc0[t][r];
        attn[((size_t)b * S_ + q0 + 16 + mh + r) * D_ + col] = (bf16)acc1[t][r];
      }
    }
  }
}

// ---------------------------------------------------------------------------
// Launch: 3 projection GEMMs -> fused attention -> output GEMM
// ---------------------------------------------------------------------------
extern "C" void kernel_launch(void* const* d_in, const int* in_sizes, int n_in,
                              void* d_out, int out_size, void* d_ws, size_t ws_size,
                              hipStream_t stream) {
  (void)in_sizes; (void)n_in; (void)out_size; (void)ws_size;

  const float* q        = (const float*)d_in[0];
  const float* k        = (const float*)d_in[1];
  const float* v        = (const float*)d_in[2];
  /* d_in[3] = mask (implicit causal, unused) */
  const int*   zero_pad = (const int*)  d_in[4];
  const float* qde      = (const float*)d_in[5];
  const float* Wk       = (const float*)d_in[6];
  const float* bk       = (const float*)d_in[7];
  const float* Wv       = (const float*)d_in[8];
  const float* bv       = (const float*)d_in[9];
  const float* Wo       = (const float*)d_in[10];
  const float* bo       = (const float*)d_in[11];
  const float* gammas   = (const float*)d_in[12];

  const size_t nBHSD = (size_t)B_ * H_ * S_ * DF_;  // 8,388,608 elems
  bf16* qh   = (bf16*)d_ws;
  bf16* kh   = qh + nBHSD;
  bf16* vhT  = kh + nBHSD;
  bf16* attn = vhT + nBHSD;  // [B, S, H*DF] bf16 (concat layout)

  const int M = B_ * S_;  // 16384
  const int N = D_;       // 512
  const int K = D_;       // 512

  dim3 gProj(M / 16, N / 64);  // (1024, 8) waves
  gemm_xwT_wmma<float, 0><<<gProj, 32, 0, stream>>>(q, Wk, bk, qh,  M, N, K);
  gemm_xwT_wmma<float, 0><<<gProj, 32, 0, stream>>>(k, Wk, bk, kh,  M, N, K);
  gemm_xwT_wmma<float, 1><<<gProj, 32, 0, stream>>>(v, Wv, bv, vhT, M, N, K);

  dim3 gAttn(S_ / 32, H_, B_);  // (16, 8, 32) waves, 32 query rows each
  akt_attention_wmma<<<gAttn, 32, 0, stream>>>(qh, kh, vhT, qde, gammas,
                                               zero_pad, attn);

  gemm_xwT_wmma<bf16, 2><<<gProj, 32, 0, stream>>>(attn, Wo, bo, (float*)d_out,
                                                   M, N, K);
}
